// FlashRobertaAttention_67757404062345
// MI455X (gfx1250) — matrix operand order repaired and checked by reading, not compiled, not run on hardware
//
#include <hip/hip_runtime.h>
#include <hip/hip_bf16.h>
#include <cstdint>

typedef __attribute__((ext_vector_type(16))) _Float16 v16h;
typedef __attribute__((ext_vector_type(8)))  _Float16 v8h;
typedef __attribute__((ext_vector_type(8)))  float    v8f;
typedef __attribute__((ext_vector_type(4)))  uint32_t u32x4;
typedef __attribute__((ext_vector_type(8)))  uint32_t u32x8;

// ---------------------------------------------------------------------------
// TDM: async 2D tile load global->LDS (dense dest), f16 elements.
// Descriptor per CDNA5 ISA 8.3/8.4 (D# group0 = 4 SGPRs, group1 = 8 SGPRs).
// Issue from ONE wave per block; follow with s_wait_tensorcnt + barrier.
// ---------------------------------------------------------------------------
__device__ __forceinline__ void tdm_load_2d_f16(const _Float16* gptr,
                                                void* lds_ptr,
                                                uint32_t tile_d0,   // elems (contig)
                                                uint32_t tile_d1,   // rows
                                                uint32_t stride_e)  // row stride, elems
{
  const uint64_t ga = (uint64_t)(uintptr_t)gptr;
  const uint32_t lds_off = (uint32_t)(uintptr_t)lds_ptr;  // LDS aperture: low 32 bits
  u32x4 g0;
  g0[0] = 1u;                                             // count=1, is_restore=0
  g0[1] = lds_off;                                        // lds_addr
  g0[2] = (uint32_t)ga;                                   // global_addr[31:0]
  g0[3] = (uint32_t)((ga >> 32) & 0x01FFFFFFu)            // global_addr[56:32]
          | 0x80000000u;                                  // type=2 ("image")
  u32x8 g1;
  g1[0] = (1u << 16);                                     // data_size=1 -> 2 bytes
  g1[1] = (tile_d0 & 0xFFFFu) << 16;                      // tensor_dim0[15:0] (== tile)
  g1[2] = (tile_d0 >> 16) | ((tile_d1 & 0xFFFFu) << 16);  // tensor_dim0 hi | tensor_dim1 lo
  g1[3] = (tile_d1 >> 16) | (tile_d0 << 16);              // tensor_dim1 hi | tile_dim0
  g1[4] = tile_d1;                                        // tile_dim1 | tile_dim2=0
  g1[5] = stride_e;                                       // tensor_dim0_stride[31:0]
  g1[6] = 0u;
  g1[7] = 0u;
  asm volatile("tensor_load_to_lds %0, %1" :: "s"(g0), "s"(g1) : "memory");
}

// ---------------------------------------------------------------------------
// f32 -> f16 convert
// ---------------------------------------------------------------------------
__global__ __launch_bounds__(256) void cvt_f32_f16(const float* __restrict__ in,
                                                   _Float16* __restrict__ out, int n) {
  int i = blockIdx.x * 256 + threadIdx.x;
  if (i < n) out[i] = (_Float16)in[i];
}

// ---------------------------------------------------------------------------
// RoPE (in-place on f16 [B*S][n_heads][64]); scale folded in (1/sqrt(64) for Q)
// ---------------------------------------------------------------------------
__global__ __launch_bounds__(256) void rope_kernel(_Float16* __restrict__ X,
                                                   int n_heads, float scale, int total) {
  int idx = blockIdx.x * 256 + threadIdx.x;
  if (idx >= total) return;
  int i    = idx & 31;          // rotation pair index 0..31
  int t    = idx >> 5;
  int head = t % n_heads;
  int bs   = t / n_heads;       // 0..B*S-1
  int s    = bs & 1023;         // position in sequence
  float inv_freq = __expf(-(float)i * 0.28782313662425572f);  // theta^(-i/32)
  float ang = (float)s * inv_freq;
  float sn, cs;
  __sincosf(ang, &sn, &cs);
  _Float16* p = X + ((size_t)bs * n_heads + head) * 64 + 2 * i;
  float x1 = (float)p[0], x2 = (float)p[1];
  p[0] = (_Float16)((x1 * cs - x2 * sn) * scale);
  p[1] = (_Float16)((x1 * sn + x2 * cs) * scale);
}

// ---------------------------------------------------------------------------
// WMMA GEMM: C[M,N] = A[M,K](f16) * B[K,N](f16) + bias (+ resid), f16 or f32 out
// block tile 128x64, K-step 32, 8 waves (each wave: 16 rows x 64 cols)
// ---------------------------------------------------------------------------
template <bool RESID, typename TOUT>
__global__ __launch_bounds__(256) void wmma_gemm(
    const _Float16* __restrict__ A, const _Float16* __restrict__ B,
    const float* __restrict__ bias, const float* __restrict__ resid,
    TOUT* __restrict__ C, int M, int N, int K) {
  __shared__ _Float16 As[128][40];  // [m][k]  (row stride 80B, 16B aligned)
  __shared__ _Float16 Bs[64][40];   // [n][k]  (transposed for contiguous K)

  const int tid  = threadIdx.x;
  const int wv   = tid >> 5;
  const int lane = tid & 31;
  const int lrow = lane & 15;
  const int lhi  = lane >> 4;
  const int n0   = blockIdx.x * 64;
  const int m0   = blockIdx.y * 128;

  v8f acc[4];
  for (int i = 0; i < 4; ++i)
    for (int r = 0; r < 8; ++r) acc[i][r] = 0.f;

  const int ksteps = K >> 5;
  for (int kt = 0; kt < ksteps; ++kt) {
    const int k0 = kt << 5;
    __syncthreads();
    // stage A tile: 128x32 halves = 512 16B chunks
    for (int i = tid; i < 512; i += 256) {
      int r = i >> 2, c = (i & 3) << 3;
      *(v8h*)&As[r][c] = *(const v8h*)&A[(size_t)(m0 + r) * K + k0 + c];
      if (kt + 1 < ksteps)
        __builtin_prefetch(&A[(size_t)(m0 + r) * K + k0 + 32 + c], 0, 0);
    }
    // stage B tile transposed: [k][n] global -> [n][k] LDS
    for (int i = tid; i < 2048; i += 256) {
      int kk = i >> 6, n = i & 63;
      Bs[n][kk] = B[(size_t)(k0 + kk) * N + n0 + n];
    }
    __syncthreads();

    // A fragment (16x32): lane&15 = row, lane>>4 selects K-half
    v16h af;
    {
      union { v16h v; v8h h[2]; } u;
      const int r = wv * 16 + lrow;
      u.h[0] = *(const v8h*)&As[r][8 * lhi];
      u.h[1] = *(const v8h*)&As[r][16 + 8 * lhi];
      af = u.v;
    }
#pragma unroll
    for (int nk = 0; nk < 4; ++nk) {
      union { v16h v; v8h h[2]; } u;
      const int n = nk * 16 + lrow;
      u.h[0] = *(const v8h*)&Bs[n][8 * lhi];
      u.h[1] = *(const v8h*)&Bs[n][16 + 8 * lhi];
      acc[nk] = __builtin_amdgcn_wmma_f32_16x16x32_f16(
          false, af, false, u.v, (short)0, acc[nk], false, false);
    }
  }

  // epilogue: C-layout row = r + 8*(lane>>4), col = lane&15
#pragma unroll
  for (int nk = 0; nk < 4; ++nk) {
    const int col = n0 + nk * 16 + lrow;
    const float bb = bias[col];
#pragma unroll
    for (int r = 0; r < 8; ++r) {
      const int row = m0 + wv * 16 + r + 8 * lhi;
      float v = acc[nk][r] + bb;
      if (RESID) v += resid[(size_t)row * N + col];
      C[(size_t)row * N + col] = (TOUT)v;
    }
  }
}

// ---------------------------------------------------------------------------
// Flash attention: grid (qtile=8, head=16, batch=8), 8 waves; each wave owns
// a 16-row q strip of the 128-row q tile. Streams 64-key tiles, online softmax.
// Q layout [B][S][16*64] f16 (pre-scaled by 1/8), K/V [B][S][8*64] f16.
// Q and K tiles are staged via the Tensor Data Mover (async, TENSORcnt).
// ---------------------------------------------------------------------------
__global__ __launch_bounds__(256) void flash_attn(
    const _Float16* __restrict__ Qh, const _Float16* __restrict__ Kh,
    const _Float16* __restrict__ Vh, _Float16* __restrict__ Oh) {
  __shared__ _Float16 Qs[128][64];   // [q][d]   (TDM dest, dense)
  __shared__ _Float16 Kt[64][64];    // [sk][d]  (TDM dest, dense)
  __shared__ _Float16 Vt[64][72];    // [d][sk]  (transposed, manual)
  __shared__ _Float16 Ps[8][16][72]; // wave-private P repack buffers

  const int tid  = threadIdx.x;
  const int wv   = tid >> 5;
  const int lane = tid & 31;
  const int lrow = lane & 15;
  const int lhi  = lane >> 4;
  const int qt = blockIdx.x, hh = blockIdx.y, bb = blockIdx.z;
  const int kvh = hh >> 1;  // GQA: 2 q-heads per kv-head

  // Q tile (128x64 from row-stride-1024 tensor) via TDM, wave 0 only
  const _Float16* Qg = Qh + ((size_t)(bb * 1024 + qt * 128) * 1024) + hh * 64;
  if (wv == 0) {
    tdm_load_2d_f16(Qg, &Qs[0][0], 64, 128, 1024);
    __builtin_amdgcn_s_wait_tensorcnt(0);
  }
  __syncthreads();

  v16h qa[2];
#pragma unroll
  for (int kk = 0; kk < 2; ++kk) {
    union { v16h v; v8h h[2]; } u;
    const int r = wv * 16 + lrow, k0 = kk * 32;
    u.h[0] = *(const v8h*)&Qs[r][k0 + 8 * lhi];
    u.h[1] = *(const v8h*)&Qs[r][k0 + 16 + 8 * lhi];
    qa[kk] = u.v;
  }

  float mrun[8], lrun[8];
  v8f oacc[4];
#pragma unroll
  for (int r = 0; r < 8; ++r) { mrun[r] = -3.0e38f; lrun[r] = 0.f; }
  for (int i = 0; i < 4; ++i)
    for (int r = 0; r < 8; ++r) oacc[i][r] = 0.f;

  const _Float16* Kbase = Kh + (size_t)bb * 1024 * 512 + kvh * 64;
  const _Float16* Vbase = Vh + (size_t)bb * 1024 * 512 + kvh * 64;

  for (int kt = 0; kt < 16; ++kt) {
    __syncthreads();  // all waves done with Kt/Vt from previous iteration
    // K tile (64x64 from row-stride-512 tensor) via TDM, wave 0 only
    if (wv == 0)
      tdm_load_2d_f16(Kbase + (size_t)(kt * 64) * 512, &Kt[0][0], 64, 64, 512);
    // V tile transposed, manual (TDM cannot transpose)
    const _Float16* Vg = Vbase + (size_t)(kt * 64) * 512;
    for (int i = tid; i < 4096; i += 256) {
      int sk = i >> 6, d = i & 63;
      Vt[d][sk] = Vg[(size_t)sk * 512 + d];
    }
    if (wv == 0) __builtin_amdgcn_s_wait_tensorcnt(0);
    __syncthreads();

    // scores S = Q * K^T  (16x64 per wave)
    v8f sacc[4];
#pragma unroll
    for (int nk = 0; nk < 4; ++nk) {
      for (int r = 0; r < 8; ++r) sacc[nk][r] = 0.f;
      const int n = nk * 16 + lrow;
#pragma unroll
      for (int kk = 0; kk < 2; ++kk) {
        union { v16h v; v8h h[2]; } u;
        const int k0 = kk * 32;
        u.h[0] = *(const v8h*)&Kt[n][k0 + 8 * lhi];
        u.h[1] = *(const v8h*)&Kt[n][k0 + 16 + 8 * lhi];
        sacc[nk] = __builtin_amdgcn_wmma_f32_16x16x32_f16(
            false, qa[kk], false, u.v, (short)0, sacc[nk], false, false);
      }
    }

    // online softmax: rows live in 16-lane halves -> xor 1/2/4/8 reduces a row
#pragma unroll
    for (int r = 0; r < 8; ++r) {
      float mx = fmaxf(fmaxf(sacc[0][r], sacc[1][r]), fmaxf(sacc[2][r], sacc[3][r]));
      mx = fmaxf(mx, __shfl_xor(mx, 1, 32));
      mx = fmaxf(mx, __shfl_xor(mx, 2, 32));
      mx = fmaxf(mx, __shfl_xor(mx, 4, 32));
      mx = fmaxf(mx, __shfl_xor(mx, 8, 32));
      const float mnew  = fmaxf(mrun[r], mx);
      const float alpha = __expf(mrun[r] - mnew);
      float rs = 0.f;
#pragma unroll
      for (int nk = 0; nk < 4; ++nk) {
        float p = __expf(sacc[nk][r] - mnew);
        sacc[nk][r] = p;
        rs += p;
      }
      rs += __shfl_xor(rs, 1, 32);
      rs += __shfl_xor(rs, 2, 32);
      rs += __shfl_xor(rs, 4, 32);
      rs += __shfl_xor(rs, 8, 32);
      lrun[r] = lrun[r] * alpha + rs;
      mrun[r] = mnew;
#pragma unroll
      for (int nd = 0; nd < 4; ++nd) oacc[nd][r] = oacc[nd][r] * alpha;
    }

    // repack P (C-layout) -> A-layout through wave-private LDS
#pragma unroll
    for (int nk = 0; nk < 4; ++nk)
#pragma unroll
      for (int r = 0; r < 8; ++r)
        Ps[wv][r + 8 * lhi][nk * 16 + lrow] = (_Float16)sacc[nk][r];
    __syncthreads();

    v16h pa[2];
#pragma unroll
    for (int kk = 0; kk < 2; ++kk) {
      union { v16h v; v8h h[2]; } u;
      const int k0 = kk * 32;
      u.h[0] = *(const v8h*)&Ps[wv][lrow][k0 + 8 * lhi];
      u.h[1] = *(const v8h*)&Ps[wv][lrow][k0 + 16 + 8 * lhi];
      pa[kk] = u.v;
    }
    // O += P * V
#pragma unroll
    for (int nd = 0; nd < 4; ++nd) {
      const int n = nd * 16 + lrow;
#pragma unroll
      for (int kk = 0; kk < 2; ++kk) {
        union { v16h v; v8h h[2]; } u;
        const int k0 = kk * 32;
        u.h[0] = *(const v8h*)&Vt[n][k0 + 8 * lhi];
        u.h[1] = *(const v8h*)&Vt[n][k0 + 16 + 8 * lhi];
        oacc[nd] = __builtin_amdgcn_wmma_f32_16x16x32_f16(
            false, pa[kk], false, u.v, (short)0, oacc[nd], false, false);
      }
    }
  }

  _Float16* Og = Oh + ((size_t)(bb * 1024 + qt * 128) * 1024) + hh * 64;
#pragma unroll
  for (int nd = 0; nd < 4; ++nd) {
    const int col = nd * 16 + lrow;
#pragma unroll
    for (int r = 0; r < 8; ++r) {
      const int row = wv * 16 + r + 8 * lhi;
      Og[(size_t)row * 1024 + col] = (_Float16)(oacc[nd][r] / lrun[r]);
    }
  }
}

// ---------------------------------------------------------------------------
// LayerNorm over D=1024 (one row per block)
// ---------------------------------------------------------------------------
__global__ __launch_bounds__(256) void ln_kernel(const float* __restrict__ resid,
                                                 const float* __restrict__ gamma,
                                                 const float* __restrict__ beta,
                                                 float* __restrict__ out) {
  const int row = blockIdx.x;
  const float* x = resid + (size_t)row * 1024;
  float* y = out + (size_t)row * 1024;
  const int tid = threadIdx.x;
  float s = 0.f, s2 = 0.f;
  for (int i = tid; i < 1024; i += 256) {
    float v = x[i];
    s += v;
    s2 += v * v;
  }
#pragma unroll
  for (int off = 16; off > 0; off >>= 1) {
    s  += __shfl_xor(s, off, 32);
    s2 += __shfl_xor(s2, off, 32);
  }
  __shared__ float ps[8], ps2[8], stats[2];
  if ((tid & 31) == 0) { ps[tid >> 5] = s; ps2[tid >> 5] = s2; }
  __syncthreads();
  if (tid == 0) {
    float ts = 0.f, ts2 = 0.f;
    for (int i = 0; i < 8; ++i) { ts += ps[i]; ts2 += ps2[i]; }
    float mu  = ts * (1.f / 1024.f);
    float var = ts2 * (1.f / 1024.f) - mu * mu;
    stats[0] = mu;
    stats[1] = rsqrtf(var + 1e-12f);
  }
  __syncthreads();
  const float mu = stats[0], rstd = stats[1];
  for (int i = tid; i < 1024; i += 256)
    y[i] = (x[i] - mu) * rstd * gamma[i] + beta[i];
}

// ---------------------------------------------------------------------------
// host launcher
// ---------------------------------------------------------------------------
extern "C" void kernel_launch(void* const* d_in, const int* in_sizes, int n_in,
                              void* d_out, int out_size, void* d_ws, size_t ws_size,
                              hipStream_t stream) {
  (void)in_sizes; (void)n_in; (void)out_size; (void)ws_size;
  const float* hidden = (const float*)d_in[0];
  const float* Wq = (const float*)d_in[1];
  const float* bq = (const float*)d_in[2];
  const float* Wk = (const float*)d_in[3];
  const float* bk = (const float*)d_in[4];
  const float* Wv = (const float*)d_in[5];
  const float* bv = (const float*)d_in[6];
  const float* Wo = (const float*)d_in[7];
  const float* bo = (const float*)d_in[8];
  const float* g  = (const float*)d_in[9];
  const float* be = (const float*)d_in[10];
  float* out = (float*)d_out;

  char* ws = (char*)d_ws;
  const size_t MB = 1024 * 1024;
  _Float16* Xh  = (_Float16*)(ws);             // 16 MB  X f16 [8192][1024]
  _Float16* Qh  = (_Float16*)(ws + 16 * MB);   // 16 MB  Q f16
  _Float16* Kh  = (_Float16*)(ws + 32 * MB);   //  8 MB  K f16
  _Float16* Vh  = (_Float16*)(ws + 40 * MB);   //  8 MB  V f16
  _Float16* Oh  = (_Float16*)(ws + 48 * MB);   // 16 MB  attn out f16
  float*    Rs  = (float*)   (ws + 64 * MB);   // 32 MB  residual f32
  _Float16* Wqh = (_Float16*)(ws + 96 * MB);   //  2 MB
  _Float16* Wkh = (_Float16*)(ws + 98 * MB);   //  1 MB
  _Float16* Wvh = (_Float16*)(ws + 99 * MB);   //  1 MB
  _Float16* Woh = (_Float16*)(ws + 100 * MB);  //  2 MB

  cvt_f32_f16<<<32768, 256, 0, stream>>>(hidden, Xh, 8388608);
  cvt_f32_f16<<<4096, 256, 0, stream>>>(Wq, Wqh, 1048576);
  cvt_f32_f16<<<2048, 256, 0, stream>>>(Wk, Wkh, 524288);
  cvt_f32_f16<<<2048, 256, 0, stream>>>(Wv, Wvh, 524288);
  cvt_f32_f16<<<4096, 256, 0, stream>>>(Wo, Woh, 1048576);

  wmma_gemm<false, _Float16><<<dim3(16, 64), 256, 0, stream>>>(
      Xh, Wqh, bq, nullptr, Qh, 8192, 1024, 1024);
  wmma_gemm<false, _Float16><<<dim3(8, 64), 256, 0, stream>>>(
      Xh, Wkh, bk, nullptr, Kh, 8192, 512, 1024);
  wmma_gemm<false, _Float16><<<dim3(8, 64), 256, 0, stream>>>(
      Xh, Wvh, bv, nullptr, Vh, 8192, 512, 1024);

  rope_kernel<<<16384, 256, 0, stream>>>(Qh, 16, 0.125f, 4194304);  // Q: fold 1/sqrt(64)
  rope_kernel<<<8192, 256, 0, stream>>>(Kh, 8, 1.0f, 2097152);

  flash_attn<<<dim3(8, 16, 8), 256, 0, stream>>>(Qh, Kh, Vh, Oh);

  wmma_gemm<true, float><<<dim3(16, 64), 256, 0, stream>>>(
      Oh, Woh, bo, hidden, Rs, 8192, 1024, 1024);

  ln_kernel<<<8192, 256, 0, stream>>>(Rs, g, be, out);
}
